// MolecularGCNWithGRU_88914412962573
// MI455X (gfx1250) — compile-verified, matrix operand order checked by reading, not run on hardware
//
#include <hip/hip_runtime.h>
#include <hip/hip_bf16.h>

// ---------------------------------------------------------------------------
// MolecularGCNWithGRU on CDNA5 (gfx1250)
//   h = feats @ W_init                       (WMMA bf16, K padded 74->96)
//   3x: agg = scatter_add(h[src] -> dst)     (f32 atomics, L2-resident)
//       x = relu(agg @ linW^T + b)           (WMMA, weights staged in LDS)
//       h = GRU(x, h)                        (6 gate GEMMs on WMMA + VALU)
// Weights are async-DMA'd to LDS once per 256-node block
// (global_load_async_to_lds_b128 + s_wait_asynccnt).
// ---------------------------------------------------------------------------

#define HD   128
#define INF  74
#define KPAD 96

typedef __attribute__((ext_vector_type(16))) __bf16 v16bf;
typedef __attribute__((ext_vector_type(8)))  float  v8f;

union Frag16 {            // 16 bf16 = 8 VGPRs (one WMMA A/B operand)
    v16bf v;
    uint4 q[2];
    unsigned short s[16];
};

__device__ __forceinline__ unsigned short f2bf(float f) {
    unsigned u = __float_as_uint(f);
    return (unsigned short)((u + 0x7FFFu + ((u >> 16) & 1u)) >> 16);
}

__device__ __forceinline__ v8f wmma_bf16(const v16bf a, const v16bf b, const v8f c) {
    return __builtin_amdgcn_wmma_f32_16x16x32_bf16(false, a, false, b, (short)0, c,
                                                   false, false);
}

// Load one K-32 chunk of an A/B operand from row-major bf16 memory (global or LDS).
// lane<16: K-offsets {0..7, 16..23}; lane>=16: {8..15, 24..31}  (koff = 0 or 8)
__device__ __forceinline__ v16bf load_frag_bf16(const unsigned short* rowptr,
                                                int s, int koff) {
    Frag16 f;
    f.q[0] = ((const uint4*)(rowptr + s * 32 + koff))[0];
    f.q[1] = ((const uint4*)(rowptr + s * 32 + koff + 16))[0];
    return f.v;
}

// Same, but sourced from f32 memory with on-the-fly bf16 conversion.
__device__ __forceinline__ v16bf load_frag_f32(const float* rowptr, int s, int koff) {
    Frag16 f;
    const float* p0 = rowptr + s * 32 + koff;
    const float* p1 = p0 + 16;
    float4 a0 = ((const float4*)p0)[0];
    float4 a1 = ((const float4*)p0)[1];
    float4 b0 = ((const float4*)p1)[0];
    float4 b1 = ((const float4*)p1)[1];
    f.s[0]  = f2bf(a0.x); f.s[1]  = f2bf(a0.y); f.s[2]  = f2bf(a0.z); f.s[3]  = f2bf(a0.w);
    f.s[4]  = f2bf(a1.x); f.s[5]  = f2bf(a1.y); f.s[6]  = f2bf(a1.z); f.s[7]  = f2bf(a1.w);
    f.s[8]  = f2bf(b0.x); f.s[9]  = f2bf(b0.y); f.s[10] = f2bf(b0.z); f.s[11] = f2bf(b0.w);
    f.s[12] = f2bf(b1.x); f.s[13] = f2bf(b1.y); f.s[14] = f2bf(b1.z); f.s[15] = f2bf(b1.w);
    return f.v;
}

__device__ __forceinline__ float sigmoidf(float x) {
    return 1.0f / (1.0f + __expf(-x));
}

// Async-DMA `total16` 16-byte chunks from uniform global base `gsrc` into LDS
// at byte offset `lds_base`, block-cooperatively. Tracked by ASYNCcnt.
__device__ __forceinline__ void async_copy_to_lds(unsigned lds_base,
                                                  const unsigned short* gsrc,
                                                  int total16, int tid, int nthr) {
    for (int i = tid; i < total16; i += nthr) {
        unsigned lds_addr = lds_base + (unsigned)i * 16u;
        unsigned voff     = (unsigned)i * 16u;
        asm volatile("global_load_async_to_lds_b128 %0, %1, %2"
                     :
                     : "v"(lds_addr), "v"(voff), "s"(gsrc)
                     : "memory");
    }
}

__device__ __forceinline__ void wait_async_all() {
    asm volatile("s_wait_asynccnt 0x0" ::: "memory");
}

// ------------------------------- converters --------------------------------

__global__ void cvt_f32_to_bf16(const float* __restrict__ in,
                                unsigned short* __restrict__ out, int n) {
    int i = blockIdx.x * blockDim.x + threadIdx.x;
    if (i < n) out[i] = f2bf(in[i]);
}

// node_feats [N,74] f32 -> [N,96] bf16 (zero-padded K)
__global__ void cvt_feats(const float* __restrict__ in,
                          unsigned short* __restrict__ out, int nrows) {
    int i = blockIdx.x * blockDim.x + threadIdx.x;
    if (i >= nrows * KPAD) return;
    int r = i / KPAD, k = i - r * KPAD;
    out[i] = (k < INF) ? f2bf(in[r * INF + k]) : (unsigned short)0;
}

// W_init [74,128] f32 -> Wt [128,96] bf16 (transposed + zero-padded K)
__global__ void cvt_winit(const float* __restrict__ W,
                          unsigned short* __restrict__ Wt) {
    int i = blockIdx.x * blockDim.x + threadIdx.x;
    if (i >= HD * KPAD) return;
    int n = i / KPAD, k = i - n * KPAD;
    Wt[i] = (k < INF) ? f2bf(W[k * HD + n]) : (unsigned short)0;
}

// ------------------------------ init transform -----------------------------
// one wave handles 16 nodes; K=96 (3 WMMA steps), 8 column tiles
__global__ __launch_bounds__(128) void init_gemm_kernel(
    const unsigned short* __restrict__ feats_bf,  // [N,96]
    const unsigned short* __restrict__ Wt,        // [128,96]
    float* __restrict__ h,                        // [N,128]
    unsigned short* __restrict__ h_bf)            // [N,128]
{
    const int lane = threadIdx.x & 31;
    const int wid  = threadIdx.x >> 5;
    const int m0   = (blockIdx.x * 4 + wid) * 16;
    const int i16  = lane & 15;
    const int hi   = lane >> 4;
    const int koff = hi * 8;

    v16bf A[3];
    const unsigned short* arow = feats_bf + (size_t)(m0 + i16) * KPAD;
#pragma unroll
    for (int s = 0; s < 3; ++s) A[s] = load_frag_bf16(arow, s, koff);

    for (int n = 0; n < 8; ++n) {
        v8f acc = {};
        const unsigned short* wrow = Wt + (size_t)(n * 16 + i16) * KPAD;
#pragma unroll
        for (int s = 0; s < 3; ++s)
            acc = wmma_bf16(A[s], load_frag_bf16(wrow, s, koff), acc);
        const int col = n * 16 + i16;
#pragma unroll
        for (int r = 0; r < 8; ++r) {
            size_t idx = (size_t)(m0 + r + hi * 8) * HD + col;
            h[idx]    = acc[r];
            h_bf[idx] = f2bf(acc[r]);
        }
    }
}

// ------------------------------ edge scatter -------------------------------
// one wave per edge; lane moves float4 (32 lanes x 4 = 128 features);
// edge indices forced scalar (s_load) via readfirstlane.
__global__ __launch_bounds__(256) void scatter_kernel(
    const float* __restrict__ h, const int* __restrict__ src,
    const int* __restrict__ dst, float* __restrict__ agg, int E)
{
    int gid  = blockIdx.x * blockDim.x + threadIdx.x;
    int e    = __builtin_amdgcn_readfirstlane(gid >> 5);
    int lane = threadIdx.x & 31;
    if (e >= E) return;
    int s = src[e];
    int d = dst[e];
    float4 v = ((const float4*)(h + (size_t)s * HD))[lane];
    float* ap = agg + (size_t)d * HD + lane * 4;
    atomicAdd(ap + 0, v.x);
    atomicAdd(ap + 1, v.y);
    atomicAdd(ap + 2, v.z);
    atomicAdd(ap + 3, v.w);
}

// ------------------------- fused GCN-linear + GRU --------------------------
// 512 threads = 16 waves, 256 nodes per block.
// Dynamic LDS layout (bytes):
//   [0,      32768)  linW   128x128 bf16
//   [32768, 131072)  Wih    384x128 bf16
//   [131072,229376)  Whh    384x128 bf16
//   [229376,294912)  x transpose slabs, 4 KB per wave
__global__ __launch_bounds__(512) void layer_kernel(
    const float* __restrict__ agg,                // [N,128] f32
    float* __restrict__ h,                        // [N,128] f32 (in/out)
    unsigned short* __restrict__ h_bf,            // [N,128] bf16 (in/out)
    const unsigned short* __restrict__ linW,      // [128,128] bf16
    const float* __restrict__ lin_b,              // [128]
    const unsigned short* __restrict__ Wih,       // [384,128] bf16
    const unsigned short* __restrict__ Whh,       // [384,128] bf16
    const float* __restrict__ bih,                // [384]
    const float* __restrict__ bhh)                // [384]
{
    extern __shared__ unsigned short smem[];
    unsigned short* lds_linW = smem;                         // 16384 shorts
    unsigned short* lds_wih  = smem + 16384;                 // 49152 shorts
    unsigned short* lds_whh  = smem + 16384 + 49152;         // 49152 shorts
    unsigned short* xslabs   = smem + 16384 + 2 * 49152;     // 16 x 2048 shorts

    const int tid  = threadIdx.x;
    const int lane = tid & 31;
    const int wid  = tid >> 5;
    const int m0   = (blockIdx.x * 16 + wid) * 16;
    const int i16  = lane & 15;                   // A-row / C-column index
    const int hi   = lane >> 4;
    const int koff = hi * 8;

    // ---- stage all weights into LDS via async DMA (overlaps A loads) ----
    const unsigned lds0 = (unsigned)(size_t)(void*)smem;
    async_copy_to_lds(lds0,                 linW, (HD * HD * 2) / 16,      tid, 512);
    async_copy_to_lds(lds0 + 32768u,        Wih,  (3 * HD * HD * 2) / 16,  tid, 512);
    async_copy_to_lds(lds0 + 131072u,       Whh,  (3 * HD * HD * 2) / 16,  tid, 512);

    // ---- A-fragments of agg (f32 -> bf16 in registers) ----
    v16bf Aagg[4];
    const float* arow = agg + (size_t)(m0 + i16) * HD;
#pragma unroll
    for (int s = 0; s < 4; ++s) Aagg[s] = load_frag_f32(arow, s, koff);

    // ---- A-fragments of h (bf16 mirror) ----
    v16bf Ah[4];
    const unsigned short* hrow = h_bf + (size_t)(m0 + i16) * HD;
#pragma unroll
    for (int s = 0; s < 4; ++s) Ah[s] = load_frag_bf16(hrow, s, koff);

    wait_async_all();
    __syncthreads();

    // ---- Phase 1: X = relu(agg @ linW^T + b) -> per-wave LDS slab ----
    unsigned short* myx = xslabs + wid * (16 * HD);
    for (int n = 0; n < 8; ++n) {
        v8f acc = {};
        const unsigned short* wrow = lds_linW + (size_t)(n * 16 + i16) * HD;
#pragma unroll
        for (int s = 0; s < 4; ++s)
            acc = wmma_bf16(Aagg[s], load_frag_bf16(wrow, s, koff), acc);
        const float bias = lin_b[n * 16 + i16];
        const int col = n * 16 + i16;
#pragma unroll
        for (int r = 0; r < 8; ++r) {
            float xv = fmaxf(acc[r] + bias, 0.0f);
            myx[(r + hi * 8) * HD + col] = f2bf(xv);
        }
    }

    // ---- Phase 2: A-fragments of x from own wave's LDS slab ----
    v16bf Ax[4];
    const unsigned short* xrow = myx + (size_t)i16 * HD;
#pragma unroll
    for (int s = 0; s < 4; ++s) Ax[s] = load_frag_bf16(xrow, s, koff);

    // ---- Phase 3: 6 gate GEMMs + GRU elementwise, write h in place ----
    for (int n = 0; n < 8; ++n) {
        v8f ir = {}, iz = {}, in_ = {}, hr = {}, hz = {}, hn = {};
        const int wr = n * 16 + i16;
        const unsigned short* wih_r = lds_wih + (size_t)(wr) * HD;
        const unsigned short* wih_z = lds_wih + (size_t)(HD + wr) * HD;
        const unsigned short* wih_n = lds_wih + (size_t)(2 * HD + wr) * HD;
        const unsigned short* whh_r = lds_whh + (size_t)(wr) * HD;
        const unsigned short* whh_z = lds_whh + (size_t)(HD + wr) * HD;
        const unsigned short* whh_n = lds_whh + (size_t)(2 * HD + wr) * HD;
#pragma unroll
        for (int s = 0; s < 4; ++s) {
            ir  = wmma_bf16(Ax[s], load_frag_bf16(wih_r, s, koff), ir);
            iz  = wmma_bf16(Ax[s], load_frag_bf16(wih_z, s, koff), iz);
            in_ = wmma_bf16(Ax[s], load_frag_bf16(wih_n, s, koff), in_);
            hr  = wmma_bf16(Ah[s], load_frag_bf16(whh_r, s, koff), hr);
            hz  = wmma_bf16(Ah[s], load_frag_bf16(whh_z, s, koff), hz);
            hn  = wmma_bf16(Ah[s], load_frag_bf16(whh_n, s, koff), hn);
        }
        const int cg = n * 16 + i16;
        const float b_ir = bih[cg],            b_hr = bhh[cg];
        const float b_iz = bih[HD + cg],       b_hz = bhh[HD + cg];
        const float b_in = bih[2 * HD + cg],   b_hn = bhh[2 * HD + cg];
#pragma unroll
        for (int r = 0; r < 8; ++r) {
            size_t idx = (size_t)(m0 + r + hi * 8) * HD + cg;
            float hv = h[idx];
            float rg = sigmoidf((ir[r] + b_ir) + (hr[r] + b_hr));
            float zg = sigmoidf((iz[r] + b_iz) + (hz[r] + b_hz));
            float ng = tanhf((in_[r] + b_in) + rg * (hn[r] + b_hn));
            float hnew = (1.0f - zg) * ng + zg * hv;
            h[idx]    = hnew;
            h_bf[idx] = f2bf(hnew);
        }
    }
}

// --------------------------------- launch ----------------------------------

extern "C" void kernel_launch(void* const* d_in, const int* in_sizes, int n_in,
                              void* d_out, int out_size, void* d_ws, size_t ws_size,
                              hipStream_t stream) {
    const float* node_feats = (const float*)d_in[0];
    const int*   src        = (const int*)d_in[1];
    const int*   dst        = (const int*)d_in[2];
    const float* W_init     = (const float*)d_in[3];
    const float* lin_W      = (const float*)d_in[4];
    const float* lin_b      = (const float*)d_in[5];
    const float* gru_Wih    = (const float*)d_in[6];
    const float* gru_Whh    = (const float*)d_in[7];
    const float* gru_bih    = (const float*)d_in[8];
    const float* gru_bhh    = (const float*)d_in[9];

    const int N = in_sizes[0] / INF;      // 131072
    const int E = in_sizes[1];            // 1048576
    const int L = 3;

    float* h = (float*)d_out;             // h lives in d_out, updated in place

    // workspace carving (agg overlays the transient bf16 feature buffer)
    char* ws = (char*)d_ws;
    float*          agg      = (float*)ws;                       // N*128*4
    unsigned short* feats_bf = (unsigned short*)ws;              // N*96*2 (pre-agg)
    size_t off = (size_t)N * HD * sizeof(float);
    unsigned short* h_bf     = (unsigned short*)(ws + off); off += (size_t)N * HD * 2;
    unsigned short* Wt_init  = (unsigned short*)(ws + off); off += (size_t)HD * KPAD * 2;
    unsigned short* linW_bf  = (unsigned short*)(ws + off); off += (size_t)L * HD * HD * 2;
    unsigned short* Wih_bf   = (unsigned short*)(ws + off); off += (size_t)L * 3 * HD * HD * 2;
    unsigned short* Whh_bf   = (unsigned short*)(ws + off); off += (size_t)L * 3 * HD * HD * 2;

    // weight / feature conversion (cheap, done every call for determinism)
    cvt_winit<<<(HD * KPAD + 255) / 256, 256, 0, stream>>>(W_init, Wt_init);
    cvt_f32_to_bf16<<<(L * HD * HD + 255) / 256, 256, 0, stream>>>(lin_W, linW_bf, L * HD * HD);
    cvt_f32_to_bf16<<<(L * 3 * HD * HD + 255) / 256, 256, 0, stream>>>(gru_Wih, Wih_bf, L * 3 * HD * HD);
    cvt_f32_to_bf16<<<(L * 3 * HD * HD + 255) / 256, 256, 0, stream>>>(gru_Whh, Whh_bf, L * 3 * HD * HD);
    cvt_feats<<<((size_t)N * KPAD + 255) / 256, 256, 0, stream>>>(node_feats, feats_bf, N);

    // h0 = feats @ W_init
    init_gemm_kernel<<<N / 64, 128, 0, stream>>>(feats_bf, Wt_init, h, h_bf);

    const size_t lds_bytes = 32768 + 2 * 98304 + 16 * 4096;   // 294912 B
    for (int l = 0; l < L; ++l) {
        hipMemsetAsync(agg, 0, (size_t)N * HD * sizeof(float), stream);
        scatter_kernel<<<(E * 32) / 256, 256, 0, stream>>>(h, src, dst, agg, E);
        layer_kernel<<<N / 256, 512, lds_bytes, stream>>>(
            agg, h, h_bf,
            linW_bf + (size_t)l * HD * HD, lin_b + (size_t)l * HD,
            Wih_bf + (size_t)l * 3 * HD * HD, Whh_bf + (size_t)l * 3 * HD * HD,
            gru_bih + (size_t)l * 3 * HD, gru_bhh + (size_t)l * 3 * HD);
    }
}
